// ProbProtoSegHead_16887811408287
// MI455X (gfx1250) — matrix operand order repaired and checked by reading, not compiled, not run on hardware
//
#include <hip/hip_runtime.h>
#include <hip/hip_bf16.h>

typedef __attribute__((ext_vector_type(16))) _Float16 v16h;
typedef __attribute__((ext_vector_type(8)))  _Float16 v8h;
typedef __attribute__((ext_vector_type(4)))  _Float16 v4h;
typedef __attribute__((ext_vector_type(8)))  float    v8f;
typedef __attribute__((ext_vector_type(4)))  float    f4;

#define NROWS 65536
#define KDIM  256
#define CMREAL 190
#define CMPAD  192
#define LAMBDA (1.0f/64.0f)
#define LN_EPS 1e-5f

union F4 { f4 v; float f[4]; };
union AF { v16h v; v8h h[2]; };

// ---------------- block reduction (256 threads = 8 waves) ----------------
__device__ inline float block_reduce_sum(float v, volatile float* red) {
#pragma unroll
  for (int m = 16; m > 0; m >>= 1) v += __shfl_xor(v, m, 32);
  int wid = threadIdx.x >> 5;
  if ((threadIdx.x & 31) == 0) red[wid] = v;
  __syncthreads();
  float r = 0.f;
#pragma unroll
  for (int i = 0; i < 8; ++i) r += red[i];
  __syncthreads();
  return r;
}

// ---------------- Kernel 1: prototype preprocessing ----------------
// pbar[cm][k] = 0.5 * sum_r l2norm(layernorm(eps_p[r]*exp(.5*pv)+proto))  (f16)
// spv[cm][k]  = sqrt(exp(proto_var))                                      (f16)
// sumpv[cm]   = sum_k exp(proto_var)                                      (f32)
__global__ void proto_prep(const float* __restrict__ proto,
                           const float* __restrict__ pvar,
                           const float* __restrict__ eps_p,
                           const float* __restrict__ lnw,
                           const float* __restrict__ lnb,
                           _Float16* __restrict__ pbar,
                           _Float16* __restrict__ spv,
                           float* __restrict__ sumpv) {
  __shared__ float red[8];
  const int cm = blockIdx.x;
  const int k  = threadIdx.x;
  if (cm >= CMREAL) {  // zero-pad columns 190..191
    pbar[cm * KDIM + k] = (_Float16)0.f;
    spv [cm * KDIM + k] = (_Float16)0.f;
    if (k == 0) sumpv[cm] = 0.f;
    return;
  }
  const float pr  = proto[cm * KDIM + k];
  const float pvl = pvar [cm * KDIM + k];
  const float w   = lnw[k], b = lnb[k];
  const float pv  = expf(pvl);
  const float sig = sqrtf(pv);          // == exp(0.5*pvl) == sqrt(exp(pvl))
  spv[cm * KDIM + k] = (_Float16)sig;
  const float tot = block_reduce_sum(pv, red);
  if (k == 0) sumpv[cm] = tot;
  float acc = 0.f;
#pragma unroll
  for (int r = 0; r < 2; ++r) {
    const float s = eps_p[(r * CMREAL + cm) * KDIM + k] * sig + pr;
    const float mean = block_reduce_sum(s, red) * (1.f / KDIM);
    const float msq  = block_reduce_sum(s * s, red) * (1.f / KDIM);
    const float t = (s - mean) * rsqrtf(msq - mean * mean + LN_EPS) * w + b;
    const float n2 = block_reduce_sum(t * t, red);
    acc += t * (1.f / fmaxf(sqrtf(n2), 1e-12f));
  }
  pbar[cm * KDIM + k] = (_Float16)(0.5f * acc);
}

// ---------------- Kernel 2: fused LN/L2 prologue + dual WMMA GEMM ----------------
// 128 threads = 4 waves; 32 rows per block; wave w owns proto-col-tiles {3w..3w+2}.
__global__ void __launch_bounds__(128)
w2_main(const float* __restrict__ x,
        const float* __restrict__ xvar,
        const float* __restrict__ eps_x,
        const float* __restrict__ lnw,
        const float* __restrict__ lnb,
        const _Float16* __restrict__ pbar,
        const _Float16* __restrict__ spv,
        const float* __restrict__ sumpv,
        float* __restrict__ out) {
  constexpr int ROWS = 32;
  constexpr int RS   = 264;  // padded LDS row stride (halves): 264*2B=528B -> bank spread
  __shared__ _Float16 R0[ROWS * RS];  // s0, later overwritten with xbar (f16)
  __shared__ _Float16 R1[ROWS * RS];  // s1
  __shared__ _Float16 R2[ROWS * RS];  // sqrt(exp(x_var))
  __shared__ float lw[KDIM], lb[KDIM];
  __shared__ float lsum[ROWS];        // sum_k exp(x_var)

  const int tid = threadIdx.x;
  const int w   = tid >> 5;   // wave id 0..3
  const int l   = tid & 31;   // lane
  const int blockBase = blockIdx.x * ROWS;

  lw[tid] = lnw[tid];  lw[tid + 128] = lnw[tid + 128];
  lb[tid] = lnb[tid];  lb[tid + 128] = lnb[tid + 128];

  // ---- phase 1: per-row reparameterize + LN stats (4 lanes per row) ----
  const int row_loc = w * 8 + (l >> 2);
  const int sub     = l & 3;                 // covers k = sub*64 .. sub*64+63
  const int n       = blockBase + row_loc;
  const f4* px  = (const f4*)(x     + (size_t)n * KDIM + sub * 64);
  const f4* pxv = (const f4*)(xvar  + (size_t)n * KDIM + sub * 64);
  const f4* pe0 = (const f4*)(eps_x + (size_t)n * KDIM + sub * 64);
  const f4* pe1 = (const f4*)(eps_x + ((size_t)NROWS + n) * KDIM + sub * 64);
  _Float16* r0 = R0 + row_loc * RS + sub * 64;
  _Float16* r1 = R1 + row_loc * RS + sub * 64;
  _Float16* r2 = R2 + row_loc * RS + sub * 64;

  float sxvs = 0.f, s0s = 0.f, s0q = 0.f, s1s = 0.f, s1q = 0.f;
#pragma unroll
  for (int i = 0; i < 16; ++i) {
    F4 vx, vv, e0, e1;
    vx.v = __builtin_nontemporal_load(px + i);   // streamed once: NT hint
    vv.v = __builtin_nontemporal_load(pxv + i);
    e0.v = __builtin_nontemporal_load(pe0 + i);
    e1.v = __builtin_nontemporal_load(pe1 + i);
    v4h hx, h0, h1;
#pragma unroll
    for (int c = 0; c < 4; ++c) {
      const float ev = expf(vv.f[c]);
      sxvs += ev;
      const float sg = sqrtf(ev);        // == exp(0.5*logvar) == sqrt(var); reused
      hx[c] = (_Float16)sg;
      const float a0 = e0.f[c] * sg + vx.f[c];
      const float a1 = e1.f[c] * sg + vx.f[c];
      s0s += a0; s0q += a0 * a0;
      s1s += a1; s1q += a1 * a1;
      h0[c] = (_Float16)a0; h1[c] = (_Float16)a1;
    }
    *(v4h*)(r2 + i * 4) = hx;
    *(v4h*)(r0 + i * 4) = h0;
    *(v4h*)(r1 + i * 4) = h1;
  }
#pragma unroll
  for (int m = 1; m <= 2; m <<= 1) {
    sxvs += __shfl_xor(sxvs, m, 32);
    s0s += __shfl_xor(s0s, m, 32);  s0q += __shfl_xor(s0q, m, 32);
    s1s += __shfl_xor(s1s, m, 32);  s1q += __shfl_xor(s1q, m, 32);
  }
  if (sub == 0) lsum[row_loc] = sxvs;
  const float mean0 = s0s * (1.f / KDIM);
  const float mean1 = s1s * (1.f / KDIM);
  const float rsq0 = rsqrtf(s0q * (1.f / KDIM) - mean0 * mean0 + LN_EPS);
  const float rsq1 = rsqrtf(s1q * (1.f / KDIM) - mean1 * mean1 + LN_EPS);

  __syncthreads();  // lw/lb visible; (s*, sxv regions are lane-private until GEMM)

  // ---- pass 2: L2 norms of layernormed samples ----
  float q0 = 0.f, q1 = 0.f;
#pragma unroll
  for (int i = 0; i < 16; ++i) {
    v4h h0 = *(const v4h*)(r0 + i * 4);
    v4h h1 = *(const v4h*)(r1 + i * 4);
#pragma unroll
    for (int c = 0; c < 4; ++c) {
      const int k = sub * 64 + i * 4 + c;
      const float t0 = ((float)h0[c] - mean0) * rsq0 * lw[k] + lb[k];
      const float t1 = ((float)h1[c] - mean1) * rsq1 * lw[k] + lb[k];
      q0 += t0 * t0; q1 += t1 * t1;
    }
  }
#pragma unroll
  for (int m = 1; m <= 2; m <<= 1) {
    q0 += __shfl_xor(q0, m, 32);
    q1 += __shfl_xor(q1, m, 32);
  }
  const float inv0 = 1.f / fmaxf(sqrtf(q0), 1e-12f);
  const float inv1 = 1.f / fmaxf(sqrtf(q1), 1e-12f);

  // ---- pass 3: xbar = 0.5*(t0/|t0| + t1/|t1|), overwrite R0 ----
#pragma unroll
  for (int i = 0; i < 16; ++i) {
    v4h h0 = *(const v4h*)(r0 + i * 4);
    v4h h1 = *(const v4h*)(r1 + i * 4);
    v4h xb;
#pragma unroll
    for (int c = 0; c < 4; ++c) {
      const int k = sub * 64 + i * 4 + c;
      const float t0 = ((float)h0[c] - mean0) * rsq0 * lw[k] + lb[k];
      const float t1 = ((float)h1[c] - mean1) * rsq1 * lw[k] + lb[k];
      xb[c] = (_Float16)(0.5f * (t0 * inv0 + t1 * inv1));
    }
    *(v4h*)(r0 + i * 4) = xb;
  }
  __syncthreads();

  // ---- phase 2: dual GEMM via v_wmma_f32_16x16x32_f16 ----
  const int r16 = l & 15;
  const int hi  = l >> 4;
  v8f accS[3][2], accC[3][2];
#pragma unroll
  for (int pt = 0; pt < 3; ++pt)
#pragma unroll
    for (int rt = 0; rt < 2; ++rt) { accS[pt][rt] = {}; accC[pt][rt] = {}; }

#pragma unroll
  for (int kb = 0; kb < 8; ++kb) {
    AF aX[2], aS[2];
#pragma unroll
    for (int rt = 0; rt < 2; ++rt) {
      // A (16x32 f16): lane holds K runs {kb*32+hi*8..+7} and {+16}
      const _Float16* pa = R0 + (rt * 16 + r16) * RS + kb * 32 + hi * 8;
      aX[rt].h[0] = *(const v8h*)pa;
      aX[rt].h[1] = *(const v8h*)(pa + 16);
      const _Float16* ps = R2 + (rt * 16 + r16) * RS + kb * 32 + hi * 8;
      aS[rt].h[0] = *(const v8h*)ps;
      aS[rt].h[1] = *(const v8h*)(ps + 16);
    }
#pragma unroll
    for (int pt = 0; pt < 3; ++pt) {
      const int j = (w * 3 + pt) * 16 + r16;   // proto column (padded to 192)
      // B (32x16 f16, column-major): lane holds K runs {kb*32+hi*8..+7} and {+16}
      const _Float16* pb = pbar + j * KDIM + kb * 32 + hi * 8;
      AF bP; bP.h[0] = *(const v8h*)pb; bP.h[1] = *(const v8h*)(pb + 16);
      const _Float16* pc = spv + j * KDIM + kb * 32 + hi * 8;
      AF bS; bS.h[0] = *(const v8h*)pc; bS.h[1] = *(const v8h*)(pc + 16);
#pragma unroll
      for (int rt = 0; rt < 2; ++rt) {
        accS[pt][rt] = __builtin_amdgcn_wmma_f32_16x16x32_f16(
            false, aX[rt].v, false, bP.v, (short)0, accS[pt][rt], false, false);
        accC[pt][rt] = __builtin_amdgcn_wmma_f32_16x16x32_f16(
            false, aS[rt].v, false, bS.v, (short)0, accC[pt][rt], false, false);
      }
    }
  }

  // ---- epilogue: out[n][cm] = -2 + 2*sim - L*(sum_xv + sum_pv) + 2*L*cross ----
#pragma unroll
  for (int pt = 0; pt < 3; ++pt) {
    const int j = (w * 3 + pt) * 16 + r16;
    if (j < CMREAL) {
      const float sp = sumpv[j];
#pragma unroll
      for (int rt = 0; rt < 2; ++rt) {
#pragma unroll
        for (int v = 0; v < 8; ++v) {
          const int nloc = rt * 16 + hi * 8 + v;   // D layout: row = hi*8 + vgpr
          const float val = -2.f + 2.f * accS[pt][rt][v]
                          - LAMBDA * (lsum[nloc] + sp)
                          + 2.f * LAMBDA * accC[pt][rt][v];
          __builtin_nontemporal_store(val, out + (size_t)(blockBase + nloc) * CMREAL + j);
        }
      }
    }
  }
}

extern "C" void kernel_launch(void* const* d_in, const int* in_sizes, int n_in,
                              void* d_out, int out_size, void* d_ws, size_t ws_size,
                              hipStream_t stream) {
  (void)in_sizes; (void)n_in; (void)out_size; (void)ws_size;
  const float* x     = (const float*)d_in[0];
  const float* xvar  = (const float*)d_in[1];
  const float* proto = (const float*)d_in[2];
  const float* pvar  = (const float*)d_in[3];
  const float* epsx  = (const float*)d_in[4];
  const float* epsp  = (const float*)d_in[5];
  const float* lnw   = (const float*)d_in[6];
  const float* lnb   = (const float*)d_in[7];
  float* out = (float*)d_out;

  _Float16* pbar = (_Float16*)d_ws;
  _Float16* spvp = pbar + CMPAD * KDIM;
  float*    sumpv = (float*)(spvp + CMPAD * KDIM);

  proto_prep<<<dim3(CMPAD), dim3(256), 0, stream>>>(proto, pvar, epsp, lnw, lnb,
                                                    pbar, spvp, sumpv);
  w2_main<<<dim3(NROWS / 32), dim3(128), 0, stream>>>(x, xvar, epsx, lnw, lnb,
                                                      pbar, spvp, sumpv, out);
}
